// StandardAttention_48928267436330
// MI455X (gfx1250) — matrix operand order repaired and checked by reading, not compiled, not run on hardware
//
#include <hip/hip_runtime.h>
#include <math.h>

typedef __attribute__((ext_vector_type(16))) _Float16 v16h;
typedef __attribute__((ext_vector_type(8)))  float    v8f;
typedef _Float16 half_t;

#define HEADS   16
#define KVHEADS 4
#define HD      128
#define SEQ     2048
#define BATCH   2
#define DMODEL  2048

union HFrag { v16h h; float4 f4[2]; };

// ---------------------------------------------------------------------------
// CDNA5 async global->LDS copy path (guarded; falls back to sync).
// Verified prototype: (int4 AS1*, int4 AS3*, imm offset, imm cpol)
// ---------------------------------------------------------------------------
typedef int v4i_vs __attribute__((vector_size(4 * sizeof(int))));
typedef __attribute__((address_space(1))) v4i_vs* as1_v4i;
typedef __attribute__((address_space(3))) v4i_vs* as3_v4i;

#if defined(__has_builtin)
# if __has_builtin(__builtin_amdgcn_global_load_async_to_lds_b128)
#  define USE_ASYNC_LDS 1
# endif
#endif

__device__ __forceinline__ void copy16_g2l(const half_t* g, half_t* l) {
#if defined(USE_ASYNC_LDS)
  __builtin_amdgcn_global_load_async_to_lds_b128((as1_v4i)g, (as3_v4i)l, 0, 0);
#else
  *(float4*)l = *(const float4*)g;
#endif
}

__device__ __forceinline__ void wait_async_copies() {
#if defined(USE_ASYNC_LDS)
# if __has_builtin(__builtin_amdgcn_s_wait_asynccnt)
  __builtin_amdgcn_s_wait_asynccnt(0);
# else
  asm volatile("s_wait_asynccnt 0" ::: "memory");
# endif
#endif
}

// ---------------------------------------------------------------------------
// GEMM: C[M,N] = A[M,K] * B[K,N]; fp32 in/out, f16 WMMA, f32 accumulate.
// N, K compile-time so strides fold into instruction offsets.
// ---------------------------------------------------------------------------
template <int N, int K>
__global__ __launch_bounds__(256)
void gemm_wmma_kernel(const float* __restrict__ A, const float* __restrict__ B,
                      float* __restrict__ C)
{
  __shared__ half_t As[128][40];   // [m][k], padded
  __shared__ half_t Bs[32][136];   // [k][n], padded
  const int tid  = threadIdx.x;
  const int wave = tid >> 5, lane = tid & 31;
  const int wm = wave >> 1, wn = wave & 1;
  const int r = lane & 15, g = lane >> 4;
  const int bm = blockIdx.y * 128, bn = blockIdx.x * 128;

  v8f acc[2][4] = {};

  for (int k0 = 0; k0 < K; k0 += 32) {
    #pragma unroll
    for (int t = 0; t < 4; ++t) {          // A tile 128x32 (float4 granule)
      int idx = tid + t * 256;
      int rr  = idx >> 3;
      int cc  = (idx & 7) << 2;
      float4 v4 = *(const float4*)(A + (size_t)(bm + rr) * K + k0 + cc);
      As[rr][cc + 0] = (half_t)v4.x; As[rr][cc + 1] = (half_t)v4.y;
      As[rr][cc + 2] = (half_t)v4.z; As[rr][cc + 3] = (half_t)v4.w;
    }
    #pragma unroll
    for (int t = 0; t < 4; ++t) {          // B tile 32x128
      int idx = tid + t * 256;
      int rr  = idx >> 5;
      int cc  = (idx & 31) << 2;
      float4 v4 = *(const float4*)(B + (size_t)(k0 + rr) * N + bn + cc);
      Bs[rr][cc + 0] = (half_t)v4.x; Bs[rr][cc + 1] = (half_t)v4.y;
      Bs[rr][cc + 2] = (half_t)v4.z; Bs[rr][cc + 3] = (half_t)v4.w;
    }
    if (k0 + 32 < K) {                     // prefetch next K-slice
      __builtin_prefetch(A + (size_t)(bm + (tid >> 1)) * K + k0 + 32 + ((tid & 1) << 4), 0, 0);
      __builtin_prefetch(B + (size_t)(k0 + 32 + (tid >> 3)) * N + bn + ((tid & 7) << 4), 0, 0);
    }
    __syncthreads();

    v16h af[2], bf[4];
    #pragma unroll
    for (int sm = 0; sm < 2; ++sm) {
      int row = wm * 32 + sm * 16 + r;
      #pragma unroll
      for (int j = 0; j < 16; ++j) {       // ISA 16-bit A 16x32 layout
        int v = j >> 1, p = j & 1;
        int kk = ((v >= 4) ? 16 : 0) + (g << 3) + ((v & 3) << 1) + p;
        af[sm][j] = As[row][kk];
      }
    }
    #pragma unroll
    for (int sn = 0; sn < 4; ++sn) {
      int col = wn * 64 + sn * 16;
      #pragma unroll
      for (int j = 0; j < 16; ++j) bf[sn][j] = Bs[((j >> 3) << 4) + (g << 3) + (j & 7)][col + r];
    }
    #pragma unroll
    for (int sm = 0; sm < 2; ++sm)
      #pragma unroll
      for (int sn = 0; sn < 4; ++sn)
        acc[sm][sn] = __builtin_amdgcn_wmma_f32_16x16x32_f16(
            false, af[sm], false, bf[sn], (short)0, acc[sm][sn], false, false);
    __syncthreads();
  }

  #pragma unroll
  for (int sm = 0; sm < 2; ++sm)
    #pragma unroll
    for (int sn = 0; sn < 4; ++sn) {
      float* cp = C + (size_t)(bm + wm * 32 + sm * 16 + (g << 3)) * N
                    + bn + wn * 64 + sn * 16 + r;
      #pragma unroll
      for (int i = 0; i < 8; ++i) cp[(size_t)i * N] = acc[sm][sn][i];
    }
}

// ---------------------------------------------------------------------------
// Q / V convert: (optional RoPE) + scale + fp32 -> f16, same row layout.
// ---------------------------------------------------------------------------
__global__ __launch_bounds__(1024)
void cvt_rope_kernel(const float* __restrict__ src, half_t* __restrict__ dst,
                     int nheads, int doRope, float scale)
{
  const int row = blockIdx.x;            // b*SEQ + s
  const int s   = row % SEQ;
  const int h   = threadIdx.x >> 6;
  const int i   = threadIdx.x & 63;
  const float* p = src + (size_t)row * nheads * HD + h * HD;
  half_t*      q = dst + (size_t)row * nheads * HD + h * HD;
  float x1 = p[i], x2 = p[i + 64];
  if (doRope) {
    float inv = powf(500000.0f, -(float)(2 * i) / (float)HD);
    float ang = (float)s * inv, c, sn;
    __sincosf(ang, &sn, &c);
    float y1 = x1 * c - x2 * sn;
    float y2 = x2 * c + x1 * sn;
    q[i]      = (half_t)(y1 * scale);
    q[i + 64] = (half_t)(y2 * scale);
  } else {
    q[i]      = (half_t)(x1 * scale);
    q[i + 64] = (half_t)(x2 * scale);
  }
}

// ---------------------------------------------------------------------------
// K convert: RoPE + fp32 -> f16, writing K TRANSPOSED: [B][kvh][feature][seq].
// Flash K tiles become contiguous along keys: async b128 staging + strided
// WMMA B-fragment reads (one key column per lane), no in-LDS transpose needed.
// ---------------------------------------------------------------------------
__global__ __launch_bounds__(256)
void cvt_rope_k_kernel(const float* __restrict__ src, half_t* __restrict__ dst)
{
  const int row = blockIdx.x;            // b*SEQ + s
  const int s   = row % SEQ;
  const int b   = row / SEQ;
  const int h   = threadIdx.x >> 6;      // kv head
  const int i   = threadIdx.x & 63;
  const float* p = src + (size_t)row * (KVHEADS * HD) + h * HD;
  float x1 = p[i], x2 = p[i + 64];
  float inv = powf(500000.0f, -(float)(2 * i) / (float)HD);
  float ang = (float)s * inv, c, sn;
  __sincosf(ang, &sn, &c);
  half_t* q = dst + ((size_t)(b * KVHEADS + h) * HD) * SEQ + s;
  q[(size_t)i * SEQ]        = (half_t)(x1 * c - x2 * sn);
  q[(size_t)(i + 64) * SEQ] = (half_t)(x2 * c + x1 * sn);
}

// ---------------------------------------------------------------------------
// Causal flash attention with GQA; f16 inputs (Q pre-scaled by 1/sqrt(d)).
// Kt global layout [B][kvh][feature][seq]; V natural [B*S][kvh*d].
// Block: 128 query rows of one (b,h); 8 waves x 16 rows; 64-key tiles.
// Mask work confined to per-wave diagonal tiles (uniform branch).
// ---------------------------------------------------------------------------
__global__ __launch_bounds__(256)
void flash_kernel(const half_t* __restrict__ Q, const half_t* __restrict__ Kt,
                  const half_t* __restrict__ V, float* __restrict__ O)
{
  __shared__ half_t Kts[HD][72];       // [feature][key]
  __shared__ half_t Vs[64][136];       // [key][feature]
  __shared__ half_t Pls[8][16][72];    // per-wave probabilities

  const int tid  = threadIdx.x;
  const int wave = tid >> 5, lane = tid & 31;
  const int r = lane & 15, g = lane >> 4;
  const int qTile = blockIdx.x;                 // 0..15
  const int bh    = blockIdx.y;                 // 0..31
  const int b = bh >> 4, h = bh & 15, kvh = h >> 2;
  const int qBase = qTile * 128;
  const int qRow  = qBase + wave * 16 + r;      // this lane's A-row

  // Q fragments: A-layout puts j=0..7 and j=8..15 at contiguous 16B chunks.
  const half_t* qrow = Q + (size_t)(b * SEQ + qRow) * (HEADS * HD) + h * HD;
  v16h qf[4];
  #pragma unroll
  for (int c = 0; c < 4; ++c) {
    HFrag fr;
    fr.f4[0] = *(const float4*)(qrow + c * 32 + (g << 3));
    fr.f4[1] = *(const float4*)(qrow + c * 32 + 16 + (g << 3));
    qf[c] = fr.h;
  }

  const half_t* ktBase = Kt + ((size_t)(b * KVHEADS + kvh) * HD) * SEQ;

  v8f oacc[8] = {};
  float mrow[8], lrow[8];
  #pragma unroll
  for (int i = 0; i < 8; ++i) { mrow[i] = -1e30f; lrow[i] = 0.0f; }

  const int qm  = qBase + wave * 16;            // wave's first query row
  const int nkb = (qBase + 128) / 64;           // causal key-tile bound
  for (int kb = 0; kb < nkb; ++kb) {
    const int kBase = kb * 64;
    #pragma unroll
    for (int t = 0; t < 4; ++t) {               // K^T tile: 128 x 64 halves
      int idx = tid + t * 256;                  // 0..1023 b128 units
      int f   = idx >> 3;                       // feature row 0..127
      int c8  = (idx & 7) << 3;                 // key col, 8-half granule
      copy16_g2l(ktBase + (size_t)f * SEQ + kBase + c8, &Kts[f][c8]);
    }
    #pragma unroll
    for (int t = 0; t < 4; ++t) {               // V tile: 64 x 128 halves
      int idx = tid + t * 256;
      int j   = idx >> 4;                       // key row 0..63
      int f8  = (idx & 15) << 3;                // feature col, 8-half granule
      copy16_g2l(V + (size_t)(b * SEQ + kBase + j) * (KVHEADS * HD) + kvh * HD + f8,
                 &Vs[j][f8]);
    }
    wait_async_copies();
    __syncthreads();

    // scores = Q * K^T over 4 key sub-tiles of 16
    v8f sacc[4] = {};
    #pragma unroll
    for (int sn = 0; sn < 4; ++sn)
      #pragma unroll
      for (int c = 0; c < 4; ++c) {
        v16h bf;                                // B[k=feature][n=key], table k order
        #pragma unroll
        for (int j = 0; j < 16; ++j) bf[j] = Kts[c * 32 + ((j >> 3) << 4) + (g << 3) + (j & 7)][sn * 16 + r];
        sacc[sn] = __builtin_amdgcn_wmma_f32_16x16x32_f16(
            false, qf[c], false, bf, (short)0, sacc[sn], false, false);
      }

    // causal mask (diagonal tiles only; wave-uniform branch) + tile row max
    float sc[4][8], tm[8];
    #pragma unroll
    for (int i = 0; i < 8; ++i) tm[i] = -1e30f;
    if (kBase + 63 > qm) {                      // diagonal / masked tile
      #pragma unroll
      for (int sn = 0; sn < 4; ++sn) {
        int kcol = kBase + sn * 16 + r;
        #pragma unroll
        for (int i = 0; i < 8; ++i) {
          float s = sacc[sn][i];
          if (kcol > qm + i + (g << 3)) s = -1e30f;
          sc[sn][i] = s;
          tm[i] = fmaxf(tm[i], s);
        }
      }
    } else {                                    // interior tile: no mask
      #pragma unroll
      for (int sn = 0; sn < 4; ++sn)
        #pragma unroll
        for (int i = 0; i < 8; ++i) {
          float s = sacc[sn][i];
          sc[sn][i] = s;
          tm[i] = fmaxf(tm[i], s);
        }
    }
    #pragma unroll
    for (int i = 0; i < 8; ++i)
      #pragma unroll
      for (int off = 8; off >= 1; off >>= 1)
        tm[i] = fmaxf(tm[i], __shfl_xor(tm[i], off, 16));

    float corr[8];
    #pragma unroll
    for (int i = 0; i < 8; ++i) {
      float mnew = fmaxf(mrow[i], tm[i]);
      corr[i] = __expf(mrow[i] - mnew);
      float rs = 0.0f;
      #pragma unroll
      for (int sn = 0; sn < 4; ++sn) {
        float p = __expf(sc[sn][i] - mnew);
        sc[sn][i] = p;
        rs += p;
      }
      #pragma unroll
      for (int off = 8; off >= 1; off >>= 1)
        rs += __shfl_xor(rs, off, 16);
      lrow[i] = lrow[i] * corr[i] + rs;
      mrow[i] = mnew;
    }

    // stash P (f16) into per-wave LDS to reshape D-layout -> A-layout
    #pragma unroll
    for (int sn = 0; sn < 4; ++sn)
      #pragma unroll
      for (int i = 0; i < 8; ++i)
        Pls[wave][i + (g << 3)][sn * 16 + r] = (half_t)sc[sn][i];

    #pragma unroll
    for (int f = 0; f < 8; ++f)
      #pragma unroll
      for (int i = 0; i < 8; ++i)
        oacc[f][i] *= corr[i];

    // O += P * V   (key dim 64 = 2 chunks of 32)
    #pragma unroll
    for (int c = 0; c < 2; ++c) {
      v16h pf;
      #pragma unroll
      for (int j = 0; j < 16; ++j) {
        int v = j >> 1, p = j & 1;
        int kk = ((v >= 4) ? 16 : 0) + (g << 3) + ((v & 3) << 1) + p;
        pf[j] = Pls[wave][r][c * 32 + kk];
      }
      #pragma unroll
      for (int f = 0; f < 8; ++f) {
        v16h bf;
        #pragma unroll
        for (int j = 0; j < 16; ++j) bf[j] = Vs[c * 32 + ((j >> 3) << 4) + (g << 3) + (j & 7)][f * 16 + r];
        oacc[f] = __builtin_amdgcn_wmma_f32_16x16x32_f16(
            false, pf, false, bf, (short)0, oacc[f], false, false);
      }
    }
    __syncthreads();
  }

  // normalize (8 reciprocals instead of 64 divides) and store [B,S,H*d]
  float rinv[8];
  #pragma unroll
  for (int i = 0; i < 8; ++i) rinv[i] = 1.0f / lrow[i];
  float* op = O + (size_t)(b * SEQ + qBase + wave * 16 + (g << 3)) * (HEADS * HD)
                + h * HD + r;
  #pragma unroll
  for (int f = 0; f < 8; ++f)
    #pragma unroll
    for (int i = 0; i < 8; ++i)
      op[i * (HEADS * HD) + f * 16] = oacc[f][i] * rinv[i];
}

// ---------------------------------------------------------------------------
extern "C" void kernel_launch(void* const* d_in, const int* in_sizes, int n_in,
                              void* d_out, int out_size, void* d_ws, size_t ws_size,
                              hipStream_t stream)
{
  (void)in_sizes; (void)n_in; (void)out_size; (void)ws_size;
  const float* x  = (const float*)d_in[0];
  const float* Wq = (const float*)d_in[1];
  const float* Wk = (const float*)d_in[2];
  const float* Wv = (const float*)d_in[3];
  const float* Wo = (const float*)d_in[4];
  float* out = (float*)d_out;

  const int    M   = BATCH * SEQ;               // 4096
  const size_t DKV = (size_t)KVHEADS * HD;      // 512

  char* ws = (char*)d_ws;
  float*  Qf  = (float*)ws;                         ws += (size_t)M * DMODEL * 4;
  float*  Kf  = (float*)ws;                         ws += (size_t)M * DKV * 4;
  float*  Vf  = (float*)ws;                         ws += (size_t)M * DKV * 4;
  half_t* Qh  = (half_t*)ws;                        ws += (size_t)M * DMODEL * 2;
  half_t* KtH = (half_t*)ws;                        ws += (size_t)M * DKV * 2;
  half_t* Vh  = (half_t*)ws;
  float*  AO  = Qf;                             // reuse Qf after conversion

  const float scale = 0.08838834764831845f;     // 1/sqrt(128)
  dim3 blk(256);
  gemm_wmma_kernel<DMODEL, DMODEL><<<dim3(DMODEL / 128, M / 128), blk, 0, stream>>>(x, Wq, Qf);
  gemm_wmma_kernel<KVHEADS * HD, DMODEL><<<dim3((KVHEADS * HD) / 128, M / 128), blk, 0, stream>>>(x, Wk, Kf);
  gemm_wmma_kernel<KVHEADS * HD, DMODEL><<<dim3((KVHEADS * HD) / 128, M / 128), blk, 0, stream>>>(x, Wv, Vf);
  cvt_rope_kernel<<<dim3(M), dim3(HEADS * 64), 0, stream>>>(Qf, Qh, HEADS, 1, scale);
  cvt_rope_k_kernel<<<dim3(M), dim3(KVHEADS * 64), 0, stream>>>(Kf, KtH);
  cvt_rope_kernel<<<dim3(M), dim3(KVHEADS * 64), 0, stream>>>(Vf, Vh, KVHEADS, 0, 1.0f);
  flash_kernel<<<dim3(SEQ / 128, BATCH * HEADS), blk, 0, stream>>>(Qh, KtH, Vh, AO);
  gemm_wmma_kernel<DMODEL, DMODEL><<<dim3(DMODEL / 128, M / 128), blk, 0, stream>>>(AO, Wo, out);
}
